// HierarchicalRSSM_19000935317873
// MI455X (gfx1250) — matrix operand order repaired and checked by reading, not compiled, not run on hardware
//
#include <hip/hip_runtime.h>

// ---------------------------------------------------------------------------
// Types / helpers for CDNA5 (gfx1250) WMMA, wave32.
// ---------------------------------------------------------------------------
typedef __attribute__((ext_vector_type(16))) __bf16 v16bf;
typedef __attribute__((ext_vector_type(8)))  float  v8f;

#define DEV __device__ __forceinline__

// Model constants
#define BATCH 64
#define TT    72      // T = INIT + SEQ + INIT
#define SEQ   64
#define INITS 4
#define BEL   1024
#define STT   256
#define ACTD  64
#define FLAT  12288
#define G3    3072    // 3*BELIEF gate width
#define OBS_OUT 50331648L  // 64*64*12288

DEV float sigmoidf_(float x){ return 1.0f / (1.0f + __expf(-x)); }
DEV float softplusf_(float x){ return (x > 20.0f) ? x : log1pf(__expf(x)); }

DEV v8f wmma_bf16(v16bf a, v16bf b, v8f c){
  return __builtin_amdgcn_wmma_f32_16x16x32_bf16(false, a, false, b, (short)0, c, false, false);
}

// A fragment: 16x32 (MxK) bf16, built from f32 row-major source.
// Lanes 0-15: row M=lane, K = k0+0..7 and k0+16..23
// Lanes 16-31: row M=lane-16, K = k0+8..15 and k0+24..31
DEV v16bf a_frag_f32(const float* A, long lda, int row0, int k0){
  int lane = threadIdx.x & 31;
  const float* p = A + (long)(row0 + (lane & 15)) * lda + k0 + ((lane >> 4) << 3);
  float4 x0 = *(const float4*)(p);
  float4 x1 = *(const float4*)(p + 4);
  float4 x2 = *(const float4*)(p + 16);
  float4 x3 = *(const float4*)(p + 20);
  v16bf f;
  f[0]=(__bf16)x0.x; f[1]=(__bf16)x0.y; f[2]=(__bf16)x0.z; f[3]=(__bf16)x0.w;
  f[4]=(__bf16)x1.x; f[5]=(__bf16)x1.y; f[6]=(__bf16)x1.z; f[7]=(__bf16)x1.w;
  f[8]=(__bf16)x2.x; f[9]=(__bf16)x2.y; f[10]=(__bf16)x2.z; f[11]=(__bf16)x2.w;
  f[12]=(__bf16)x3.x; f[13]=(__bf16)x3.y; f[14]=(__bf16)x3.z; f[15]=(__bf16)x3.w;
  return f;
}

// Same layout, source already bf16: two 16-byte loads, no conversion.
DEV v16bf a_frag_bf16(const __bf16* A, long lda, int row0, int k0){
  int lane = threadIdx.x & 31;
  const __bf16* p = A + (long)(row0 + (lane & 15)) * lda + k0 + ((lane >> 4) << 3);
  union { uint4 u[2]; v16bf v; } t;
  t.u[0] = *(const uint4*)(p);
  t.u[1] = *(const uint4*)(p + 16);
  return t.v;
}

// With per-row scale (rs may be null); rss is row stride in rs.
DEV v16bf a_frag_f32_s(const float* A, long lda, int row0, int k0,
                       const float* rs, long rss){
  int lane = threadIdx.x & 31;
  int r = row0 + (lane & 15);
  float s = rs ? rs[(long)r * rss] : 1.0f;
  const float* p = A + (long)r * lda + k0 + ((lane >> 4) << 3);
  float4 x0 = *(const float4*)(p);
  float4 x1 = *(const float4*)(p + 4);
  float4 x2 = *(const float4*)(p + 16);
  float4 x3 = *(const float4*)(p + 20);
  v16bf f;
  f[0]=(__bf16)(x0.x*s); f[1]=(__bf16)(x0.y*s); f[2]=(__bf16)(x0.z*s); f[3]=(__bf16)(x0.w*s);
  f[4]=(__bf16)(x1.x*s); f[5]=(__bf16)(x1.y*s); f[6]=(__bf16)(x1.z*s); f[7]=(__bf16)(x1.w*s);
  f[8]=(__bf16)(x2.x*s); f[9]=(__bf16)(x2.y*s); f[10]=(__bf16)(x2.z*s); f[11]=(__bf16)(x2.w*s);
  f[12]=(__bf16)(x3.x*s); f[13]=(__bf16)(x3.y*s); f[14]=(__bf16)(x3.z*s); f[15]=(__bf16)(x3.w*s);
  return f;
}

// B fragment: 32x16 (KxN) bf16 from weight stored transposed (N x K) row-major.
// Lanes 0-15: col N=lane, K=k0+0..7 and k0+16..23.  Lanes 16-31: col N=lane-16, K=k0+8..15 and k0+24..31.
DEV v16bf b_frag_bf16(const __bf16* Wt, long ldk, int n0, int k0){
  int lane = threadIdx.x & 31;
  const __bf16* p = Wt + (long)(n0 + (lane & 15)) * ldk + k0 + ((lane >> 4) << 3);
  union { uint4 u[2]; v16bf v; } tmp;
  tmp.u[0] = *(const uint4*)(p);
  tmp.u[1] = *(const uint4*)(p + 16);
  return tmp.v;
}

// Accumulator element (v) -> (row, col): row = row0 + v + 8*(lane>=16), col = col0 + (lane&15)

// ---------------------------------------------------------------------------
// Generic GEMM: out = act(A @ Wt^T + bias). WG tile 64 x 128 (wave = 16x64, 4 accs)
// ABF: A is bf16 (else f32). OBF: out is bf16 (else f32).
// act: 0 none, 1 relu, 2 sigmoid. remap: 0 identity; 1: r=t*64+b -> out row b*64+t.
// Grid (M/64, N/128), block 256.
// ---------------------------------------------------------------------------
template<int ABF, int OBF>
__global__ __launch_bounds__(256)
void gemm_bias_act(const void* __restrict__ Av, long lda,
                   const __bf16* __restrict__ Wt, const float* __restrict__ bias,
                   void* __restrict__ outv, int N, int K, int act, int remap)
{
  const float*  Af = (const float*)Av;
  const __bf16* Ab = (const __bf16*)Av;
  int wid = threadIdx.x >> 5;
  int row0 = blockIdx.x * 64 + (wid & 3) * 16;
  int col0 = blockIdx.y * 128 + (wid >> 2) * 64;
  v8f acc[4];
#pragma unroll
  for (int q = 0; q < 4; ++q) acc[q] = (v8f){};
  for (int k0 = 0; k0 < K; k0 += 32){
    if (k0 + 32 < K)  // pull next B tile toward L2/L0 (global_prefetch_b8)
      __builtin_prefetch((const char*)Wt + ((long)col0 * K + k0 + 32) * 2, 0, 0);
    v16bf a = ABF ? a_frag_bf16(Ab, lda, row0, k0) : a_frag_f32(Af, lda, row0, k0);
#pragma unroll
    for (int q = 0; q < 4; ++q)
      acc[q] = wmma_bf16(a, b_frag_bf16(Wt, K, col0 + 16*q, k0), acc[q]);
  }
  int cbase = threadIdx.x & 15;
#pragma unroll
  for (int v = 0; v < 8; ++v){
    int r = row0 + v + (((threadIdx.x >> 4) & 1) << 3);
    long orow = r;
    if (remap){ orow = (long)(r & 63) * 64 + (r >> 6); }
#pragma unroll
    for (int q = 0; q < 4; ++q){
      int c = col0 + 16*q + cbase;
      float x = acc[q][v] + bias[c];
      if (act == 1) x = fmaxf(x, 0.0f);
      else if (act == 2) x = sigmoidf_(x);
      if (OBF) ((__bf16*)outv)[orow * (long)N + c] = (__bf16)x;
      else     ((float*)outv)[orow * (long)N + c] = x;
    }
  }
}

// ---------------------------------------------------------------------------
// Concat GEMM (M=64): f32 A sources; writes f32 and/or bf16 (either may be null).
// Grid N/64, block 256.
// ---------------------------------------------------------------------------
__global__ __launch_bounds__(256)
void cat_gemm_kernel(const float* __restrict__ A0, long l0,
                     const float* __restrict__ A1, long l1, int ksplit, int K,
                     const __bf16* __restrict__ Wt, const float* __restrict__ bias,
                     float* __restrict__ outf, __bf16* __restrict__ outb, long ldo)
{
  int wid = threadIdx.x >> 5;
  int row0 = (wid & 3) * 16;
  int col0 = blockIdx.x * 64 + (wid >> 2) * 32;
  v8f acc0 = {}; v8f acc1 = {};
  for (int k0 = 0; k0 < K; k0 += 32){
    v16bf a = (k0 < ksplit) ? a_frag_f32(A0, l0, row0, k0)
                            : a_frag_f32(A1, l1, row0, k0 - ksplit);
    acc0 = wmma_bf16(a, b_frag_bf16(Wt, K, col0,      k0), acc0);
    acc1 = wmma_bf16(a, b_frag_bf16(Wt, K, col0 + 16, k0), acc1);
  }
  int cbase = threadIdx.x & 15;
#pragma unroll
  for (int v = 0; v < 8; ++v){
    int r = row0 + v + (((threadIdx.x >> 4) & 1) << 3);
#pragma unroll
    for (int f2 = 0; f2 < 2; ++f2){
      int c = col0 + 16*f2 + cbase;
      float x = (f2 ? acc1[v] : acc0[v]) + bias[c];
      if (outf) outf[(long)r * ldo + c] = x;
      if (outb) outb[(long)r * ldo + c] = (__bf16)x;
    }
  }
}

// ---------------------------------------------------------------------------
// GRU core: accumulates all three gate GEMMs for a 16-row x 32-col slice.
// x = concat(x0,x1) (K=Kx, split; bf16 if XBF), h (64 x 1024 f32, optional row scale).
// Wit: (3072 x Kx) bf16, Wht: (3072 x 1024) bf16.
// aR/aZ get gi+gh sums; aIN gets x-part of n; aHN gets h-part of n.
// ---------------------------------------------------------------------------
template<int XBF>
DEV void gru_core(const void* x0, long l0, const void* x1, long l1, int ksplit, int Kx,
                  const float* h, const float* rs, long rss,
                  const __bf16* Wit, const __bf16* Wht,
                  int row0, int n0,
                  v8f (&aR)[2], v8f (&aZ)[2], v8f (&aIN)[2], v8f (&aHN)[2])
{
#pragma unroll
  for (int f2 = 0; f2 < 2; ++f2){ aR[f2] = (v8f){}; aZ[f2] = (v8f){}; aIN[f2] = (v8f){}; aHN[f2] = (v8f){}; }
  for (int k0 = 0; k0 < Kx; k0 += 32){
    const void* xp = (k0 < ksplit) ? x0 : x1;
    long  xl = (k0 < ksplit) ? l0 : l1;
    int   kk = (k0 < ksplit) ? k0 : k0 - ksplit;
    v16bf a = XBF ? a_frag_bf16((const __bf16*)xp, xl, row0, kk)
                  : a_frag_f32((const float*)xp, xl, row0, kk);
#pragma unroll
    for (int f2 = 0; f2 < 2; ++f2){
      int nc = n0 + 16*f2;
      aR[f2]  = wmma_bf16(a, b_frag_bf16(Wit, Kx, nc,          k0), aR[f2]);
      aZ[f2]  = wmma_bf16(a, b_frag_bf16(Wit, Kx, BEL + nc,    k0), aZ[f2]);
      aIN[f2] = wmma_bf16(a, b_frag_bf16(Wit, Kx, 2*BEL + nc,  k0), aIN[f2]);
    }
  }
  for (int k0 = 0; k0 < BEL; k0 += 32){
    v16bf a = a_frag_f32_s(h, BEL, row0, k0, rs, rss);
#pragma unroll
    for (int f2 = 0; f2 < 2; ++f2){
      int nc = n0 + 16*f2;
      aR[f2]  = wmma_bf16(a, b_frag_bf16(Wht, BEL, nc,         k0), aR[f2]);
      aZ[f2]  = wmma_bf16(a, b_frag_bf16(Wht, BEL, BEL + nc,   k0), aZ[f2]);
      aHN[f2] = wmma_bf16(a, b_frag_bf16(Wht, BEL, 2*BEL + nc, k0), aHN[f2]);
    }
  }
}

DEV float gru_elem(float sr, float sz, float sin_, float shn,
                   const float* bi, const float* bh, int cc, float hval){
  float rg = sigmoidf_(sr + bi[cc] + bh[cc]);
  float zg = sigmoidf_(sz + bi[BEL + cc] + bh[BEL + cc]);
  float ng = tanhf(sin_ + bi[2*BEL + cc] + rg * (shn + bh[2*BEL + cc]));
  return (1.0f - zg) * ng + zg * hval;
}

// ---------------------------------------------------------------------------
// Forward scan step: blocks 0..15 -> apf, 16..31 -> opf (h scaled by copy gate).
// x = enc_bf (bf16).
// ---------------------------------------------------------------------------
__global__ __launch_bounds__(256)
void fwd_gru_kernel(const __bf16* __restrict__ enc, const float* __restrict__ bnd,
                    const float* __restrict__ afc_in, float* __restrict__ afc_out,
                    float* __restrict__ af_all,
                    const float* __restrict__ ofc_in, float* __restrict__ ofc_out,
                    float* __restrict__ of_all,
                    const __bf16* __restrict__ apfWit, const __bf16* __restrict__ apfWht,
                    const float* __restrict__ apf_bi, const float* __restrict__ apf_bh,
                    const __bf16* __restrict__ opfWit, const __bf16* __restrict__ opfWht,
                    const float* __restrict__ opf_bi, const float* __restrict__ opf_bh,
                    int t)
{
  int g = blockIdx.x;
  int sub = g >> 4;
  int wid = threadIdx.x >> 5;
  int row0 = (wid & 3) * 16;
  int n0 = (g & 15) * 64 + (wid >> 2) * 32;
  const float* h; const float* rs = nullptr; long rss = 0;
  const __bf16 *Wit, *Wht; const float *bi, *bh;
  float *cout, *hist;
  if (sub == 0){ h = afc_in; Wit = apfWit; Wht = apfWht; bi = apf_bi; bh = apf_bh;
                 cout = afc_out; hist = af_all + (long)t * BATCH * BEL; }
  else { h = ofc_in; rs = bnd + t*2 + 1; rss = TT*2;
         Wit = opfWit; Wht = opfWht; bi = opf_bi; bh = opf_bh;
         cout = ofc_out; hist = of_all + (long)t * BATCH * BEL; }
  v8f aR[2], aZ[2], aIN[2], aHN[2];
  gru_core<1>(enc + (long)t*BEL, (long)TT*BEL, nullptr, 0, 1 << 30, BEL,
              h, rs, rss, Wit, Wht, row0, n0, aR, aZ, aIN, aHN);
#pragma unroll
  for (int f2 = 0; f2 < 2; ++f2){
    int cc = n0 + 16*f2 + (threadIdx.x & 15);
#pragma unroll
    for (int v = 0; v < 8; ++v){
      int rr = row0 + v + (((threadIdx.x >> 4) & 1) << 3);
      float hv = h[(long)rr*BEL + cc];
      if (rs) hv *= rs[(long)rr*rss];
      float gv = gru_elem(aR[f2][v], aZ[f2][v], aIN[f2][v], aHN[f2][v], bi, bh, cc, hv);
      cout[(long)rr*BEL + cc] = gv;
      hist[(long)rr*BEL + cc] = gv;
    }
  }
}

// Backward scan step: ab = gru(e_t, carry); ab_all[t]=ab; carry' = copy_t * ab.
__global__ __launch_bounds__(256)
void bwd_gru_kernel(const __bf16* __restrict__ enc, const float* __restrict__ bnd,
                    const float* __restrict__ c_in, float* __restrict__ c_out,
                    float* __restrict__ ab_all,
                    const __bf16* __restrict__ Wit, const __bf16* __restrict__ Wht,
                    const float* __restrict__ bi, const float* __restrict__ bh, int t)
{
  int wid = threadIdx.x >> 5;
  int row0 = (wid & 3) * 16;
  int n0 = blockIdx.x * 64 + (wid >> 2) * 32;
  v8f aR[2], aZ[2], aIN[2], aHN[2];
  gru_core<1>(enc + (long)t*BEL, (long)TT*BEL, nullptr, 0, 1 << 30, BEL,
              c_in, nullptr, 0, Wit, Wht, row0, n0, aR, aZ, aIN, aHN);
  float* hist = ab_all + (long)t * BATCH * BEL;
#pragma unroll
  for (int f2 = 0; f2 < 2; ++f2){
    int cc = n0 + 16*f2 + (threadIdx.x & 15);
#pragma unroll
    for (int v = 0; v < 8; ++v){
      int rr = row0 + v + (((threadIdx.x >> 4) & 1) << 3);
      float hv = c_in[(long)rr*BEL + cc];
      float gv = gru_elem(aR[f2][v], aZ[f2][v], aIN[f2][v], aHN[f2][v], bi, bh, cc, hv);
      float cp = bnd[((long)rr*TT + t)*2 + 1];
      hist[(long)rr*BEL + cc] = gv;
      c_out[(long)rr*BEL + cc] = cp * gv;
    }
  }
}

// Main scan: abstract-belief update. x = [a_s(256), act_prev(64)] K=320 (f32).
__global__ __launch_bounds__(256)
void m1_kernel(const float* __restrict__ a_s, const float* __restrict__ act,
               const float* __restrict__ ab_in, float* __restrict__ ab_out,
               const float* __restrict__ af_all, const float* __restrict__ bnd,
               const __bf16* __restrict__ Wit, const __bf16* __restrict__ Wht,
               const float* __restrict__ bi, const float* __restrict__ bh, int t)
{
  int wid = threadIdx.x >> 5;
  int row0 = (wid & 3) * 16;
  int n0 = blockIdx.x * 64 + (wid >> 2) * 32;
  v8f aR[2], aZ[2], aIN[2], aHN[2];
  gru_core<0>(a_s, STT, act + (long)(INITS - 1 + t)*ACTD, (long)TT*ACTD, STT, STT + ACTD,
              ab_in, nullptr, 0, Wit, Wht, row0, n0, aR, aZ, aIN, aHN);
  const float* afp = af_all + (long)(INITS - 1 + t) * BATCH * BEL;
  const float* bndp = bnd + (INITS + t) * 2;
#pragma unroll
  for (int f2 = 0; f2 < 2; ++f2){
    int cc = n0 + 16*f2 + (threadIdx.x & 15);
#pragma unroll
    for (int v = 0; v < 8; ++v){
      int rr = row0 + v + (((threadIdx.x >> 4) & 1) << 3);
      float hv = ab_in[(long)rr*BEL + cc];
      float upd = gru_elem(aR[f2][v], aZ[f2][v], aIN[f2][v], aHN[f2][v], bi, bh, cc, hv);
      float rd = bndp[(long)rr*TT*2 + 0];
      float cp = bndp[(long)rr*TT*2 + 1];
      float abn = (t == 0) ? afp[(long)rr*BEL + cc] : (cp * hv + rd * upd);
      ab_out[(long)rr*BEL + cc] = abn;
    }
  }
}

// Main scan: observation-belief update. x = [o_s(256), feat(1024)] K=1280 (f32).
// ob_out = copy * gru(...) + read * feat
__global__ __launch_bounds__(256)
void m4_kernel(const float* __restrict__ o_s, const float* __restrict__ feat,
               const float* __restrict__ ob_in, float* __restrict__ ob_out,
               const float* __restrict__ bnd,
               const __bf16* __restrict__ Wit, const __bf16* __restrict__ Wht,
               const float* __restrict__ bi, const float* __restrict__ bh, int t)
{
  int wid = threadIdx.x >> 5;
  int row0 = (wid & 3) * 16;
  int n0 = blockIdx.x * 64 + (wid >> 2) * 32;
  v8f aR[2], aZ[2], aIN[2], aHN[2];
  gru_core<0>(o_s, STT, feat, BEL, STT, STT + BEL,
              ob_in, nullptr, 0, Wit, Wht, row0, n0, aR, aZ, aIN, aHN);
  const float* bndp = bnd + (INITS + t) * 2;
#pragma unroll
  for (int f2 = 0; f2 < 2; ++f2){
    int cc = n0 + 16*f2 + (threadIdx.x & 15);
#pragma unroll
    for (int v = 0; v < 8; ++v){
      int rr = row0 + v + (((threadIdx.x >> 4) & 1) << 3);
      float hv = ob_in[(long)rr*BEL + cc];
      float gv = gru_elem(aR[f2][v], aZ[f2][v], aIN[f2][v], aHN[f2][v], bi, bh, cc, hv);
      float rd = bndp[(long)rr*TT*2 + 0];
      float cp = bndp[(long)rr*TT*2 + 1];
      ob_out[(long)rr*BEL + cc] = cp * gv + rd * feat[(long)rr*BEL + cc];
    }
  }
}

// Latent head: (m, softplus(s)+1e-4) from concat input; update state s.
// mode 0: s = copy*s + read*(m + sd*eps);  mode 1: s = m + sd*eps.
__global__ __launch_bounds__(256)
void lat_kernel(const float* __restrict__ A0, long l0,
                const float* __restrict__ A1, long l1, int ksplit, int K,
                const __bf16* __restrict__ Wmt, const float* __restrict__ bm,
                const __bf16* __restrict__ Wst, const float* __restrict__ bs,
                const float* __restrict__ epsb, long ldeps,
                const float* __restrict__ bndp,
                float* __restrict__ s, int mode)
{
  int wid = threadIdx.x >> 5;
  int row0 = (wid & 3) * 16;
  int n0 = blockIdx.x * 64 + (wid >> 2) * 32;
  v8f am[2]; v8f av[2];
#pragma unroll
  for (int f2 = 0; f2 < 2; ++f2){ am[f2] = (v8f){}; av[f2] = (v8f){}; }
  for (int k0 = 0; k0 < K; k0 += 32){
    v16bf a = (k0 < ksplit) ? a_frag_f32(A0, l0, row0, k0)
                            : a_frag_f32(A1, l1, row0, k0 - ksplit);
#pragma unroll
    for (int f2 = 0; f2 < 2; ++f2){
      am[f2] = wmma_bf16(a, b_frag_bf16(Wmt, K, n0 + 16*f2, k0), am[f2]);
      av[f2] = wmma_bf16(a, b_frag_bf16(Wst, K, n0 + 16*f2, k0), av[f2]);
    }
  }
#pragma unroll
  for (int f2 = 0; f2 < 2; ++f2){
    int cc = n0 + 16*f2 + (threadIdx.x & 15);
#pragma unroll
    for (int v = 0; v < 8; ++v){
      int rr = row0 + v + (((threadIdx.x >> 4) & 1) << 3);
      float m = am[f2][v] + bm[cc];
      float sd = softplusf_(av[f2][v] + bs[cc]) + 1e-4f;
      float val = m + sd * epsb[(long)rr*ldeps + cc];
      long si = (long)rr*STT + cc;
      if (mode == 0){
        float rd = bndp[(long)rr*TT*2 + 0];
        float cp = bndp[(long)rr*TT*2 + 1];
        s[si] = cp * s[si] + rd * val;
      } else {
        s[si] = val;
      }
    }
  }
}

// ---------------------------------------------------------------------------
// Tiny N=2 head (bf16 A, f32 W): out[orow*2 + {0,1}] = A[r] . W(:,{0,1}) + b
// ---------------------------------------------------------------------------
__global__ void out2_kernel(const __bf16* __restrict__ A, long lda,
                            const float* __restrict__ W, const float* __restrict__ b2,
                            float* __restrict__ out, int M, int K, int remap)
{
  int r = blockIdx.x * blockDim.x + threadIdx.x;
  if (r >= M) return;
  const __bf16* a = A + (long)r * lda;
  float s0 = b2[0], s1 = b2[1];
  for (int k = 0; k < K; ++k){
    float av = (float)a[k];
    s0 += av * W[k*2];
    s1 += av * W[k*2 + 1];
  }
  long orow = r;
  if (remap) orow = (long)(r & 63) * 64 + (r >> 6);
  out[orow*2] = s0; out[orow*2 + 1] = s1;
}

// Boundary: straight-through hard one-hot of log-softmax + force mask.
__global__ void boundary_kernel(const float* __restrict__ pla,
                                float* __restrict__ bnd, float* __restrict__ bout)
{
  int i = blockIdx.x * blockDim.x + threadIdx.x;
  if (i >= BATCH * TT) return;
  int b = i / TT, t = i % TT;
  float la0 = pla[i*2], la1 = pla[i*2 + 1];
  float mx = fmaxf(la0, la1);
  float lse = mx + __logf(__expf(la0 - mx) + __expf(la1 - mx));
  float p0 = __expf(la0 - lse), p1 = __expf(la1 - lse);
  float h0 = (la0 >= la1) ? 1.0f : 0.0f;
  float b0 = (h0 - p0) + p0, b1 = ((1.0f - h0) - p1) + p1;
  if (t <= INITS || t >= TT - INITS){ b0 = 1.0f; b1 = 0.0f; }
  bnd[i*2] = b0; bnd[i*2 + 1] = b1;
  if (t >= INITS && t < INITS + SEQ){
    long o = ((long)b * SEQ + (t - INITS)) * 2;
    bout[o] = b0; bout[o + 1] = b1;
  }
}

// Transpose+convert: W (K x N) f32 -> Wt (N x K) bf16. block (32,8), grid (K/32, N/32).
__global__ void transpose_conv_kernel(const float* __restrict__ W, __bf16* __restrict__ Wt,
                                      int K, int N)
{
  __shared__ float tile[32][33];
  int kb = blockIdx.x * 32, nb = blockIdx.y * 32;
  int tx = threadIdx.x, ty = threadIdx.y;
#pragma unroll
  for (int i = 0; i < 32; i += 8)
    tile[ty + i][tx] = W[(long)(kb + ty + i) * N + nb + tx];
  __syncthreads();
#pragma unroll
  for (int i = 0; i < 32; i += 8)
    Wt[(long)(nb + ty + i) * K + kb + tx] = (__bf16)tile[tx][ty + i];
}

// Flat f32 -> bf16 convert (n divisible by 4).
__global__ void conv_bf16_kernel(const float* __restrict__ in, __bf16* __restrict__ out, long n)
{
  long i = ((long)blockIdx.x * blockDim.x + threadIdx.x) * 4;
  if (i >= n) return;
  float4 x = *(const float4*)(in + i);
  out[i]   = (__bf16)x.x;
  out[i+1] = (__bf16)x.y;
  out[i+2] = (__bf16)x.z;
  out[i+3] = (__bf16)x.w;
}

// ---------------------------------------------------------------------------
// Host orchestration
// ---------------------------------------------------------------------------
extern "C" void kernel_launch(void* const* d_in, const int* in_sizes, int n_in,
                              void* d_out, int out_size, void* d_ws, size_t ws_size,
                              hipStream_t stream)
{
  (void)in_sizes; (void)n_in; (void)out_size; (void)ws_size;
  const float* obs  = (const float*)d_in[0];
  const float* act  = (const float*)d_in[1];
  const float* encW = (const float*)d_in[2];  const float* encB = (const float*)d_in[3];
  const float* decW = (const float*)d_in[4];  const float* decB = (const float*)d_in[5];
  const float* pbW[3] = {(const float*)d_in[6], (const float*)d_in[8], (const float*)d_in[10]};
  const float* pbB[3] = {(const float*)d_in[7], (const float*)d_in[9], (const float*)d_in[11]};
  const float* pboW = (const float*)d_in[12]; const float* pboB = (const float*)d_in[13];
  const float* prbW = (const float*)d_in[14]; const float* prbB = (const float*)d_in[15];
  const float* afW  = (const float*)d_in[16]; const float* afB  = (const float*)d_in[17];
  const float* ofW  = (const float*)d_in[18]; const float* ofB  = (const float*)d_in[19];
  const float* uaWi = (const float*)d_in[20]; const float* uaWh = (const float*)d_in[21];
  const float* ua_bi= (const float*)d_in[22]; const float* ua_bh= (const float*)d_in[23];
  const float* uoWi = (const float*)d_in[24]; const float* uoWh = (const float*)d_in[25];
  const float* uo_bi= (const float*)d_in[26]; const float* uo_bh= (const float*)d_in[27];
  const float* apfWi= (const float*)d_in[28]; const float* apfWh= (const float*)d_in[29];
  const float* apf_bi=(const float*)d_in[30]; const float* apf_bh=(const float*)d_in[31];
  const float* apbWi= (const float*)d_in[32]; const float* apbWh= (const float*)d_in[33];
  const float* apb_bi=(const float*)d_in[34]; const float* apb_bh=(const float*)d_in[35];
  const float* opfWi= (const float*)d_in[36]; const float* opfWh= (const float*)d_in[37];
  const float* opf_bi=(const float*)d_in[38]; const float* opf_bh=(const float*)d_in[39];
  const float* paWm = (const float*)d_in[40]; const float* pa_bm = (const float*)d_in[41];
  const float* paWs = (const float*)d_in[42]; const float* pa_bs = (const float*)d_in[43];
  const float* poWm = (const float*)d_in[44]; const float* po_bm = (const float*)d_in[45];
  const float* poWs = (const float*)d_in[46]; const float* po_bs = (const float*)d_in[47];
  const float* epsA = (const float*)d_in[48];
  const float* epsO = (const float*)d_in[49];

  // ---- workspace carve ----
  size_t off = 0;
  char* wsb = (char*)d_ws;
  auto take = [&](size_t bytes)->char*{
    char* p = wsb + off;
    off += (bytes + 255) & ~(size_t)255;
    return p;
  };
  __bf16* encWt = (__bf16*)take((size_t)FLAT*BEL*2);
  __bf16* decWt = (__bf16*)take((size_t)FLAT*BEL*2);
  __bf16* pbWt[3]; for (int i=0;i<3;++i) pbWt[i] = (__bf16*)take((size_t)BEL*BEL*2);
  __bf16* afWt  = (__bf16*)take((size_t)BEL*(BEL+STT)*2);
  __bf16* ofWt  = (__bf16*)take((size_t)BEL*(BEL+STT)*2);
  __bf16* uaWit = (__bf16*)take((size_t)G3*(STT+ACTD)*2);
  __bf16* uaWht = (__bf16*)take((size_t)G3*BEL*2);
  __bf16* uoWit = (__bf16*)take((size_t)G3*(STT+BEL)*2);
  __bf16* uoWht = (__bf16*)take((size_t)G3*BEL*2);
  __bf16* apfWit= (__bf16*)take((size_t)G3*BEL*2);
  __bf16* apfWht= (__bf16*)take((size_t)G3*BEL*2);
  __bf16* apbWit= (__bf16*)take((size_t)G3*BEL*2);
  __bf16* apbWht= (__bf16*)take((size_t)G3*BEL*2);
  __bf16* opfWit= (__bf16*)take((size_t)G3*BEL*2);
  __bf16* opfWht= (__bf16*)take((size_t)G3*BEL*2);
  __bf16* paWmt = (__bf16*)take((size_t)STT*2*BEL*2);
  __bf16* paWst = (__bf16*)take((size_t)STT*2*BEL*2);
  __bf16* poWmt = (__bf16*)take((size_t)STT*2*BEL*2);
  __bf16* poWst = (__bf16*)take((size_t)STT*2*BEL*2);

  const size_t NTB = (size_t)BATCH * TT * BEL;   // 4718592
  __bf16* obs_bf  = (__bf16*)take((size_t)BATCH*TT*FLAT*2);
  __bf16* enc_bf  = (__bf16*)take(NTB*2);
  __bf16* pbh0    = (__bf16*)take(NTB*2);
  __bf16* pbh1    = (__bf16*)take(NTB*2);
  __bf16* ofeatsb = (__bf16*)take((size_t)SEQ*BATCH*BEL*2);
  float* pla    = (float*)take((size_t)BATCH*TT*2*4);
  float* bndb   = (float*)take((size_t)BATCH*TT*2*4);
  float* af_all = (float*)take(NTB*4);
  float* of_all = (float*)take(NTB*4);
  float* ab_all = (float*)take(NTB*4);
  float* afc[2]; float* ofc[2]; float* bwc[2]; float* abb[2]; float* obb[2];
  for (int i=0;i<2;++i){ afc[i]=(float*)take(65536*4); ofc[i]=(float*)take(65536*4);
                         bwc[i]=(float*)take(65536*4); abb[i]=(float*)take(65536*4);
                         obb[i]=(float*)take(65536*4); }
  float* a_s  = (float*)take((size_t)BATCH*STT*4);
  float* o_s  = (float*)take((size_t)BATCH*STT*4);
  float* feat = (float*)take((size_t)BATCH*BEL*4);

  // ---- weight transpose + bf16 convert ----
  auto T2 = [&](const float* W, __bf16* Wt, int K, int N){
    transpose_conv_kernel<<<dim3(K/32, N/32), dim3(32,8), 0, stream>>>(W, Wt, K, N);
  };
  T2(encW, encWt, FLAT, BEL);
  T2(decW, decWt, BEL, FLAT);
  for (int i=0;i<3;++i) T2(pbW[i], pbWt[i], BEL, BEL);
  T2(afW, afWt, BEL+STT, BEL);
  T2(ofW, ofWt, BEL+STT, BEL);
  T2(uaWi, uaWit, STT+ACTD, G3);  T2(uaWh, uaWht, BEL, G3);
  T2(uoWi, uoWit, STT+BEL, G3);   T2(uoWh, uoWht, BEL, G3);
  T2(apfWi, apfWit, BEL, G3);     T2(apfWh, apfWht, BEL, G3);
  T2(apbWi, apbWit, BEL, G3);     T2(apbWh, apbWht, BEL, G3);
  T2(opfWi, opfWit, BEL, G3);     T2(opfWh, opfWht, BEL, G3);
  T2(paWm, paWmt, 2*BEL, STT);    T2(paWs, paWst, 2*BEL, STT);
  T2(poWm, poWmt, 2*BEL, STT);    T2(poWs, poWst, 2*BEL, STT);

  // ---- obs f32 -> bf16 (L2-resident A for the encoder GEMM) ----
  {
    long n = (long)BATCH * TT * FLAT;
    conv_bf16_kernel<<<(unsigned)((n/4 + 255)/256),256,0,stream>>>(obs, obs_bf, n);
  }

  // ---- zero carries ----
  hipMemsetAsync(afc[0], 0, 65536*4, stream);
  hipMemsetAsync(ofc[0], 0, 65536*4, stream);
  hipMemsetAsync(bwc[0], 0, 65536*4, stream);
  hipMemsetAsync(abb[0], 0, 65536*4, stream);
  hipMemsetAsync(obb[0], 0, 65536*4, stream);
  hipMemsetAsync(a_s,  0, (size_t)BATCH*STT*4, stream);
  hipMemsetAsync(o_s,  0, (size_t)BATCH*STT*4, stream);

  // ---- encoder + boundary MLP (bf16 activations end-to-end) ----
  gemm_bias_act<1,1><<<dim3(72,8),256,0,stream>>>(obs_bf, (long)FLAT, encWt, encB, enc_bf, BEL, FLAT, 1, 0);
  gemm_bias_act<1,1><<<dim3(72,8),256,0,stream>>>(enc_bf, (long)BEL, pbWt[0], pbB[0], pbh0, BEL, BEL, 1, 0);
  gemm_bias_act<1,1><<<dim3(72,8),256,0,stream>>>(pbh0,   (long)BEL, pbWt[1], pbB[1], pbh1, BEL, BEL, 1, 0);
  gemm_bias_act<1,1><<<dim3(72,8),256,0,stream>>>(pbh1,   (long)BEL, pbWt[2], pbB[2], pbh0, BEL, BEL, 1, 0);
  out2_kernel<<<(BATCH*TT+255)/256,256,0,stream>>>(pbh0, (long)BEL, pboW, pboB, pla, BATCH*TT, BEL, 0);
  boundary_kernel<<<(BATCH*TT+255)/256,256,0,stream>>>(pla, bndb, (float*)d_out + OBS_OUT + (long)BATCH*SEQ*2);

  // ---- forward scan (apf + opf fused per step) ----
  for (int t = 0; t < TT; ++t)
    fwd_gru_kernel<<<32,256,0,stream>>>(enc_bf, bndb,
        afc[t&1], afc[(t+1)&1], af_all, ofc[t&1], ofc[(t+1)&1], of_all,
        apfWit, apfWht, apf_bi, apf_bh, opfWit, opfWht, opf_bi, opf_bh, t);

  // ---- backward scan ----
  for (int i = 0; i < TT; ++i){
    int t = TT - 1 - i;
    bwd_gru_kernel<<<16,256,0,stream>>>(enc_bf, bndb, bwc[i&1], bwc[(i+1)&1], ab_all,
                                        apbWit, apbWht, apb_bi, apb_bh, t);
  }

  // ---- main hierarchical scan ----
  for (int t = 0; t < SEQ; ++t){
    const float* abin = abb[t&1]; float* about = abb[(t+1)&1];
    const float* obin = obb[t&1]; float* obout = obb[(t+1)&1];
    m1_kernel<<<16,256,0,stream>>>(a_s, act, abin, about, af_all, bndb,
                                   uaWit, uaWht, ua_bi, ua_bh, t);
    lat_kernel<<<4,256,0,stream>>>(af_all + (size_t)(INITS-1+t)*65536, (long)BEL,
                                   ab_all + (size_t)(INITS+t)*65536, (long)BEL, BEL, 2*BEL,
                                   paWmt, pa_bm, paWst, pa_bs,
                                   epsA + (size_t)t*STT, (long)SEQ*STT,
                                   bndb + (INITS+t)*2, a_s, 0);
    cat_gemm_kernel<<<16,256,0,stream>>>(about, (long)BEL, a_s, (long)STT, BEL, BEL+STT,
                                         afWt, afB, feat, (__bf16*)nullptr, (long)BEL);
    m4_kernel<<<16,256,0,stream>>>(o_s, feat, obin, obout, bndb,
                                   uoWit, uoWht, uo_bi, uo_bh, t);
    lat_kernel<<<4,256,0,stream>>>(of_all + (size_t)(INITS+t)*65536, (long)BEL,
                                   feat, (long)BEL, BEL, 2*BEL,
                                   poWmt, po_bm, poWst, po_bs,
                                   epsO + (size_t)t*STT, (long)SEQ*STT,
                                   bndb + (INITS+t)*2, o_s, 1);
    cat_gemm_kernel<<<16,256,0,stream>>>(obout, (long)BEL, o_s, (long)STT, BEL, BEL+STT,
                                         ofWt, ofB, (float*)nullptr,
                                         ofeatsb + (size_t)t*65536, (long)BEL);
  }

  // ---- decoder + prior boundary head ----
  gemm_bias_act<1,0><<<dim3(64,96),256,0,stream>>>(ofeatsb, (long)BEL, decWt, decB,
                                                   (float*)d_out, FLAT, BEL, 2, 1);
  out2_kernel<<<(SEQ*BATCH+255)/256,256,0,stream>>>(ofeatsb, (long)BEL, prbW, prbB,
                                                    (float*)d_out + OBS_OUT, SEQ*BATCH, BEL, 1);
}
